// LinW_Attention_Module_18348100288825
// MI455X (gfx1250) — matrix operand order repaired and checked by reading, not compiled, not run on hardware
//
#include <hip/hip_runtime.h>
#include <hip/hip_bf16.h>

typedef __attribute__((ext_vector_type(16))) _Float16 v16h;
typedef __attribute__((ext_vector_type(8)))  _Float16 v8h;
typedef __attribute__((ext_vector_type(8)))  float    v8f;

#define BATCH 16
#define DIM   512
#define HEADS 8

// ---------------------------------------------------------------------------
// Threefry-2x32-20 (exact JAX PRNG reproduction)
// ---------------------------------------------------------------------------
__device__ __forceinline__ unsigned rotl32(unsigned v, int n) {
    return (v << n) | (v >> (32 - n));
}

__device__ void threefry2x32(unsigned k0, unsigned k1,
                             unsigned x0, unsigned x1,
                             unsigned& o0, unsigned& o1) {
    const unsigned ks2 = k0 ^ k1 ^ 0x1BD11BDAu;
    x0 += k0; x1 += k1;
    x0 += x1; x1 = rotl32(x1, 13); x1 ^= x0;
    x0 += x1; x1 = rotl32(x1, 15); x1 ^= x0;
    x0 += x1; x1 = rotl32(x1, 26); x1 ^= x0;
    x0 += x1; x1 = rotl32(x1,  6); x1 ^= x0;
    x0 += k1; x1 += ks2 + 1u;
    x0 += x1; x1 = rotl32(x1, 17); x1 ^= x0;
    x0 += x1; x1 = rotl32(x1, 29); x1 ^= x0;
    x0 += x1; x1 = rotl32(x1, 16); x1 ^= x0;
    x0 += x1; x1 = rotl32(x1, 24); x1 ^= x0;
    x0 += ks2; x1 += k0 + 2u;
    x0 += x1; x1 = rotl32(x1, 13); x1 ^= x0;
    x0 += x1; x1 = rotl32(x1, 15); x1 ^= x0;
    x0 += x1; x1 = rotl32(x1, 26); x1 ^= x0;
    x0 += x1; x1 = rotl32(x1,  6); x1 ^= x0;
    x0 += k0; x1 += k1 + 3u;
    x0 += x1; x1 = rotl32(x1, 17); x1 ^= x0;
    x0 += x1; x1 = rotl32(x1, 29); x1 ^= x0;
    x0 += x1; x1 = rotl32(x1, 16); x1 ^= x0;
    x0 += x1; x1 = rotl32(x1, 24); x1 ^= x0;
    x0 += k1; x1 += ks2 + 4u;
    x0 += x1; x1 = rotl32(x1, 13); x1 ^= x0;
    x0 += x1; x1 = rotl32(x1, 15); x1 ^= x0;
    x0 += x1; x1 = rotl32(x1, 26); x1 ^= x0;
    x0 += x1; x1 = rotl32(x1,  6); x1 ^= x0;
    x0 += ks2; x1 += k0 + 5u;
    o0 = x0; o1 = x1;
}

struct RngKeys { unsigned kp0, kp1, km0, km1, kr10, kr11, kr20, kr21; };

// jax.random.key(42) -> (0,42); split(key,4): bits = threefry(key, iota(8)),
// counter pairs (i, i+4); keys = bits.reshape(4,2)
__device__ RngKeys derive_keys() {
    RngKeys r;
    unsigned b0,b1,b2,b3,b4,b5,b6,b7;
    threefry2x32(0u, 42u, 0u, 4u, b0, b4);
    threefry2x32(0u, 42u, 1u, 5u, b1, b5);
    threefry2x32(0u, 42u, 2u, 6u, b2, b6);
    threefry2x32(0u, 42u, 3u, 7u, b3, b7);
    r.kp0 = b0; r.kp1 = b1;
    r.km0 = b2; r.km1 = b3;
    r.kr10 = b4; r.kr11 = b5;
    r.kr20 = b6; r.kr21 = b7;
    return r;
}

__device__ __forceinline__ float u01(unsigned bits) {
    return __uint_as_float((bits >> 9) | 0x3f800000u) - 1.0f;
}

__device__ unsigned uniform_bits(unsigned k0, unsigned k1, unsigned e, unsigned half) {
    unsigned p = (e < half) ? e : e - half;
    unsigned o0, o1;
    threefry2x32(k0, k1, p, p + half, o0, o1);
    return (e < half) ? o0 : o1;
}

// randint(key,(16,8),0,512): span 512 pow2 -> bits & 511 (exact)
__device__ int randint512(unsigned k0, unsigned k1, int e) {
    return (int)(uniform_bits(k0, k1, (unsigned)e, 64u) & 511u);
}

// ---------------------------------------------------------------------------
// Global -> LDS tile loader, writing fragment-order ("swizzled") layout.
// Tile is 64 rows x 32 k, source row-major with row stride 512 (contiguous k).
// LDS layout: lds[tile(4)][laneHalf(2)][laneRow(16)][half(16)], so one lane's
// v16h fragment is 32 contiguous bytes  -> 2x ds_load_b128, no repacking.
//   A half mapping (16x32 f16 A): lanes 0-15 hold K {0..7,16..23};
//                                 lanes16-31 hold K {8..15,24..31}
//     8-run kq in {0,8,16,24}: hs = (kq>>3)&1, h0 = (kq>>4)*8
//   B half mapping (32x16 f16 B): same as A: lanes 0-15 hold K {0..7,16..23};
//     lanes16-31 hold K {8..15,24..31}
// ---------------------------------------------------------------------------
template<bool SRC_F16, bool IS_A>
__device__ __forceinline__ void load_tile_swz(const void* __restrict__ src,
                                              long base, int row0, int k0,
                                              _Float16* lds, int tid) {
    #pragma unroll
    for (int r = tid; r < 256; r += 128) {       // 256 8-elem runs, 2 per thread
        const int row = r >> 2;                  // 0..63
        const int kq  = (r & 3) << 3;            // 0,8,16,24
        const long g  = base + (long)(row0 + row) * DIM + (k0 + kq);
        _Float16 h[8];
        if (SRC_F16) {
            const v8h val = *(const v8h*)((const _Float16*)src + g);
            #pragma unroll
            for (int x = 0; x < 8; ++x) h[x] = val[x];
        } else {
            const float4 f0 = *(const float4*)((const float*)src + g);
            const float4 f1 = *(const float4*)((const float*)src + g + 4);
            h[0] = (_Float16)f0.x; h[1] = (_Float16)f0.y;
            h[2] = (_Float16)f0.z; h[3] = (_Float16)f0.w;
            h[4] = (_Float16)f1.x; h[5] = (_Float16)f1.y;
            h[6] = (_Float16)f1.z; h[7] = (_Float16)f1.w;
        }
        const int tile = row >> 4, rr = row & 15;
        const int hs = (kq >> 3) & 1;
        const int h0 = (kq >> 4) << 3;
        v8h pack;
        #pragma unroll
        for (int x = 0; x < 8; ++x) pack[x] = h[x];
        *(v8h*)&lds[(((tile * 2 + hs) * 16 + rr) * 16) + h0] = pack;
    }
}

// ---------------------------------------------------------------------------
// Tiled WMMA f16 GEMM (M=N=K=512):  Out = alpha * A x B^T_src + bias
//   A source:  Asrc[m*512 + k]            (row-major, contiguous k)
//   B source:  Bsrc[n*512 + k]            (contiguous k, i.e. "n-major")
//   STORE_T:   Out[n*512 + m]  (8-half vector stores)  else Out[m*512 + n]
// 64x64 block tile, 4 waves, each wave 2x2 of 16x16 WMMA tiles, K step 32.
// ---------------------------------------------------------------------------
template<bool SRC_F16, bool HAS_BIAS, bool STORE_T>
__global__ __launch_bounds__(128)
void wmma_gemm_kernel(const void* __restrict__ Aptr,
                      const void* __restrict__ Bptr,
                      const float* __restrict__ bias,
                      _Float16* __restrict__ Out,
                      float alpha, long aBatchStride, long bBatchStride) {
    constexpr int MT = 64, NT = 64, KT = 32;
    __shared__ _Float16 As[4 * 2 * 16 * 16];     // 2048 halves, fragment order
    __shared__ _Float16 Bs[4 * 2 * 16 * 16];

    const int b    = blockIdx.z;
    const int m0   = blockIdx.y * MT;
    const int n0   = blockIdx.x * NT;
    const int tid  = threadIdx.x;
    const int lane = tid & 31;
    const int wave = tid >> 5;
    const int hs   = lane >> 4;
    const int lid  = lane & 15;
    const int ta0  = (wave >> 1) * 2;            // wave's A tile base (of 4)
    const int tb0  = (wave & 1) * 2;             // wave's B tile base (of 4)

    v8f zero = {};
    v8f acc[2][2];
    acc[0][0] = zero; acc[0][1] = zero; acc[1][0] = zero; acc[1][1] = zero;

    const long aBase = (long)b * aBatchStride;
    const long bBase = (long)b * bBatchStride;

    // hoisted per-column bias (constant over k loop and over v)
    float bv[2] = {0.f, 0.f};
    if (HAS_BIAS) {
        #pragma unroll
        for (int tc = 0; tc < 2; ++tc)
            bv[tc] = bias[n0 + (tb0 + tc) * 16 + lid];
    }

    for (int k0 = 0; k0 < DIM; k0 += KT) {
        load_tile_swz<SRC_F16, true >(Aptr, aBase, m0, k0, As, tid);
        load_tile_swz<SRC_F16, false>(Bptr, bBase, n0, k0, Bs, tid);
        if (k0 + KT < DIM) {   // prefetch next k-tile (global_prefetch_b8)
            const int row = (tid >> 2) & 63;
            __builtin_prefetch((const char*)Aptr + (aBase + (long)(m0 + row) * DIM + k0 + KT) * (SRC_F16 ? 2 : 4), 0, 1);
            __builtin_prefetch((const char*)Bptr + (bBase + (long)(n0 + row) * DIM + k0 + KT) * (SRC_F16 ? 2 : 4), 0, 1);
        }
        __syncthreads();

        v16h af[2], bf[2];
        #pragma unroll
        for (int t = 0; t < 2; ++t) {
            af[t] = *(const v16h*)&As[(((ta0 + t) * 2 + hs) * 16 + lid) * 16];
            bf[t] = *(const v16h*)&Bs[(((tb0 + t) * 2 + hs) * 16 + lid) * 16];
        }
        #pragma unroll
        for (int tr = 0; tr < 2; ++tr)
            #pragma unroll
            for (int tc = 0; tc < 2; ++tc)
                acc[tr][tc] = __builtin_amdgcn_wmma_f32_16x16x32_f16(
                    false, af[tr], false, bf[tc],
                    (short)0, acc[tr][tc], false, false);
        __syncthreads();
    }

    // Epilogue.  C/D layout: VGPR v -> M = v + 8*hs, N = lane&15.
    const long oBase = (long)b * DIM * DIM;
    #pragma unroll
    for (int tr = 0; tr < 2; ++tr)
        #pragma unroll
        for (int tc = 0; tc < 2; ++tc) {
            const int ng = n0 + (tb0 + tc) * 16 + lid;
            if (STORE_T) {
                // consecutive v -> consecutive m: one 16B store of 8 halves
                const int mg0 = m0 + (ta0 + tr) * 16 + 8 * hs;
                v8h d;
                #pragma unroll
                for (int v = 0; v < 8; ++v)
                    d[v] = (_Float16)(acc[tr][tc][v] * alpha + bv[tc]);
                *(v8h*)&Out[oBase + (long)ng * DIM + mg0] = d;
            } else {
                #pragma unroll
                for (int v = 0; v < 8; ++v) {
                    const int mg = m0 + (ta0 + tr) * 16 + v + 8 * hs;
                    Out[oBase + (long)mg * DIM + ng] =
                        (_Float16)(acc[tr][tc][v] * alpha + bv[tc]);
                }
            }
        }
}

// ---------------------------------------------------------------------------
// Vp = V @ WV_w^T + WV_b    (16x512, trivial)
// ---------------------------------------------------------------------------
__global__ __launch_bounds__(256)
void vp_kernel(const float* __restrict__ V, const float* __restrict__ Ww,
               const float* __restrict__ Wb, float* __restrict__ Vp) {
    int idx = blockIdx.x * 256 + threadIdx.x;     // 8192 = 16*512
    int b = idx >> 9, i = idx & 511;
    const float* v = V + b * DIM;
    const float* w = Ww + (long)i * DIM;
    float s = 0.f;
    for (int d = 0; d < DIM; ++d) s = fmaf(v[d], w[d], s);
    Vp[idx] = s + Wb[i];
}

// ---------------------------------------------------------------------------
// Mutate/crossover + softmax + attn.Vp.  One block per (b,i) row, 8 heads.
// ---------------------------------------------------------------------------
__global__ __launch_bounds__(512)
void attn_kernel(const _Float16* __restrict__ S, const float* __restrict__ Vp,
                 float* __restrict__ out3) {
    const int i = blockIdx.x, b = blockIdx.y;
    const int j = threadIdx.x;                 // 0..511
    const int lane = j & 31, wid = j >> 5;

    __shared__ float redA[16], redB[16];
    __shared__ float shMax;

    const RngKeys rk = derive_keys();
    // p = uniform(kp, ()): odd-size counter -> pair (0,0), take out0
    unsigned p0, p1;
    threefry2x32(rk.kp0, rk.kp1, 0u, 0u, p0, p1);
    const bool mutate = (u01(p0) <= 0.6f);

    const float vj = Vp[b * DIM + j];
    const float base_s = (float)S[((long)b * DIM + i) * DIM + j];
    const int   c = DIM - (int)(DIM * 0.3);    // 359; tail = columns [c, 512)

    for (int h = 0; h < HEADS; ++h) {
        float v;
        if (mutate) {
            unsigned e = ((unsigned)((b * HEADS + h) * DIM + i)) * DIM + (unsigned)j;
            unsigned bits = uniform_bits(rk.km0, rk.km1, e, 16777216u);
            float m = fmaxf(0.7f, u01(bits) * 0.6f + 0.7f);
            v = base_s * m;
        } else {
            v = base_s;
            int e = b * HEADS + h;
            int r1 = randint512(rk.kr10, rk.kr11, e);
            int r2 = randint512(rk.kr20, rk.kr21, e);
            if (j >= c && r1 != r2) {
                if (i == r1)      v = (float)S[((long)b * DIM + r2) * DIM + j];
                else if (i == r2) v = (float)S[((long)b * DIM + r1) * DIM + j];
            }
        }
        // row max
        float m = v;
        for (int s = 16; s > 0; s >>= 1) m = fmaxf(m, __shfl_xor(m, s, 32));
        if (lane == 0) redA[wid] = m;
        __syncthreads();
        if (j == 0) {
            float t = redA[0];
            for (int w = 1; w < 16; ++w) t = fmaxf(t, redA[w]);
            shMax = t;
        }
        __syncthreads();
        // sum(exp) and sum(exp * Vp)
        float ex = __expf(v - shMax);
        float s1 = ex, s2 = ex * vj;
        for (int s = 16; s > 0; s >>= 1) {
            s1 += __shfl_xor(s1, s, 32);
            s2 += __shfl_xor(s2, s, 32);
        }
        if (lane == 0) { redA[wid] = s1; redB[wid] = s2; }
        __syncthreads();
        if (j == 0) {
            float t1 = 0.f, t2 = 0.f;
            for (int w = 0; w < 16; ++w) { t1 += redA[w]; t2 += redB[w]; }
            out3[((long)b * HEADS + h) * DIM + i] = t2 / t1;
        }
        __syncthreads();
    }
}

// ---------------------------------------------------------------------------
// final = out3.reshape(16,4096) @ WO_w^T + WO_b
// ---------------------------------------------------------------------------
__global__ __launch_bounds__(256)
void out_proj_kernel(const float* __restrict__ out3, const float* __restrict__ WOw,
                     const float* __restrict__ WOb, float* __restrict__ out) {
    int idx = blockIdx.x * 256 + threadIdx.x;   // 8192 = 16*512
    int b = idx >> 9, o = idx & 511;
    const float* x = out3 + (long)b * (HEADS * DIM);
    const float* w = WOw + (long)o * (HEADS * DIM);
    float s = 0.f;
    for (int cidx = 0; cidx < HEADS * DIM; ++cidx) s = fmaf(x[cidx], w[cidx], s);
    out[idx] = s + WOb[o];
}

// ---------------------------------------------------------------------------
extern "C" void kernel_launch(void* const* d_in, const int* in_sizes, int n_in,
                              void* d_out, int out_size, void* d_ws, size_t ws_size,
                              hipStream_t stream) {
    const float* Q   = (const float*)d_in[0];
    const float* K   = (const float*)d_in[1];
    const float* V   = (const float*)d_in[2];
    const float* WQw = (const float*)d_in[3];
    const float* WQb = (const float*)d_in[4];
    const float* WKw = (const float*)d_in[5];
    const float* WKb = (const float*)d_in[6];
    const float* WVw = (const float*)d_in[7];
    const float* WVb = (const float*)d_in[8];
    const float* WOw = (const float*)d_in[9];
    const float* WOb = (const float*)d_in[10];
    float* outp = (float*)d_out;

    // workspace layout (~25.5 MB)
    char* ws = (char*)d_ws;
    _Float16* QpT  = (_Float16*)(ws + 0);          // 8 MiB: QpT[i][j] = Qp[j][i]
    _Float16* KpT  = (_Float16*)(ws + 8388608);    // 8 MiB: KpT[k][j] = Kp[j][k]
    _Float16* S    = (_Float16*)(ws + 16777216);   // 8 MiB: S[i][k] (heads share)
    float*    Vp   = (float*)(ws + 25165824);      // 32 KiB
    float*    out3 = (float*)(ws + 25198592);      // 256 KiB (B,H,A)

    const long MAT = (long)DIM * DIM;
    dim3 gGemm(DIM / 64, DIM / 64, BATCH), bGemm(128);

    // Qp^T = (Q @ WQ^T + bq)^T     A=Q (f32, row-major), B=WQ_w ([n][k])
    wmma_gemm_kernel<false, true, true><<<gGemm, bGemm, 0, stream>>>(
        Q, WQw, WQb, QpT, 1.0f, MAT, 0L);
    // Kp^T = (K @ WK^T + bk)^T
    wmma_gemm_kernel<false, true, true><<<gGemm, bGemm, 0, stream>>>(
        K, WKw, WKb, KpT, 1.0f, MAT, 0L);
    // Vp (f32)
    vp_kernel<<<32, 256, 0, stream>>>(V, WVw, WVb, Vp);
    // S[i][k] = (1/sqrt(8)) * sum_j Qp[j][i] Kp[j][k]
    //         = QpT (A, row-major over j) x KpT (B, [n=k][k=j])
    wmma_gemm_kernel<true, false, false><<<gGemm, bGemm, 0, stream>>>(
        QpT, KpT, nullptr, S, 0.35355339059327373f, MAT, MAT);
    // RNG perturbation + softmax + attn . Vp
    attn_kernel<<<dim3(DIM, BATCH), 512, 0, stream>>>(S, Vp, out3);
    // output projection
    out_proj_kernel<<<32, 256, 0, stream>>>(out3, WOw, WOb, outp);
}
